// Attention_13735305412836
// MI455X (gfx1250) — compile-verified, matrix operand order checked by reading, not run on hardware
//
#include <hip/hip_runtime.h>
#include <hip/hip_bf16.h>

// ---------------------------------------------------------------------------
// Multi-head attention for MI455X (gfx1250), wave32 + v_wmma_f32_16x16x32_f16
//   b=2, n=2048, dim=1024, heads=16, head_dim=64, inner=1024
// Pipeline:
//   1) f32->f16 convert of q,k,v ; weights converted + transposed to [N][K]
//   2) WMMA GEMM projections -> Qh (pre-scaled), Kh in [b,h,n,64]; V -> [b,h,64,n]
//   3) flash-attention: block = 4 waves = 64 query rows of one (b,h);
//      K/V tiles double-buffered in LDS via global_load_async_to_lds_b128
//      (ASYNCcnt), online softmax, P transposed through LDS (DScnt)
//   4) WMMA GEMM output projection -> f32 out
// ---------------------------------------------------------------------------

typedef __attribute__((ext_vector_type(16))) _Float16 v16h;
typedef __attribute__((ext_vector_type(8)))  _Float16 v8h;
typedef __attribute__((ext_vector_type(8)))  float    v8f;

constexpr int BSZ   = 2;
constexpr int SEQ   = 2048;
constexpr int DIM   = 1024;
constexpr int HEADS = 16;
constexpr int HD    = 64;
constexpr int INNER = HEADS * HD;   // 1024
constexpr int ROWS  = BSZ * SEQ;    // 4096
constexpr int KD    = 1024;         // contraction dim for both GEMM stages

__device__ __forceinline__ v16h cat16(v8h lo, v8h hi) {
    return __builtin_shufflevector(lo, hi, 0,1,2,3,4,5,6,7,8,9,10,11,12,13,14,15);
}

__device__ __forceinline__ v8f wmma_f16(v16h a, v16h b, v8f c) {
    // v_wmma_f32_16x16x32_f16  D = A(16x32) * B(32x16) + C(16x16 f32)
    return __builtin_amdgcn_wmma_f32_16x16x32_f16(
        /*neg_a=*/false, a, /*neg_b=*/false, b,
        /*c_mod=*/(short)0, c, /*reuse_a=*/false, /*reuse_b=*/false);
}

// gfx1250 async global->LDS DMA, 16B per active lane, tracked by ASYNCcnt.
// VDST operand = LDS byte address (low 32 bits of the flat shared pointer;
// HW adds the wave's LDS_BASE).
__device__ __forceinline__ void async_copy16(_Float16* lptr, const _Float16* gptr) {
    uint32_t la = (uint32_t)(uintptr_t)lptr;
    uint64_t ga = (uint64_t)(uintptr_t)gptr;
    asm volatile("global_load_async_to_lds_b128 %0, %1, off"
                 :: "v"(la), "v"(ga) : "memory");
}
__device__ __forceinline__ void async_wait0() {
    asm volatile("s_wait_asynccnt 0" ::: "memory");
}

// ---------------------------------------------------------------------------
// elementwise f32 -> f16
__global__ void k_cvt_f16(const float* __restrict__ src,
                          _Float16* __restrict__ dst, int n) {
    int i = blockIdx.x * 256 + threadIdx.x;
    if (i < n) dst[i] = (_Float16)src[i];
}

// W[K][N] f32 -> Wt[N][K] f16  (so B-fragments are contiguous row loads)
__global__ void k_w_transpose(const float* __restrict__ W,
                              _Float16* __restrict__ Wt, int K, int N) {
    int i = blockIdx.x * 256 + threadIdx.x;
    if (i < K * N) {
        int n = i / K;
        int k = i - n * K;
        Wt[i] = (_Float16)W[(size_t)k * N + n];
    }
}

// ---------------------------------------------------------------------------
// GEMM: C[M,N] = A[M,K] * Bt[N,K]^T ; one wave computes a 16x64 tile.
// mode 0/1: out[b,h,i,d] f16 (scaled)   mode 2: out[b,h,d,i] f16 (V transposed)
__global__ __launch_bounds__(128)
void k_proj(const _Float16* __restrict__ A, const _Float16* __restrict__ Bt,
            _Float16* __restrict__ out, float scale, int mode) {
    const int lane = threadIdx.x & 31;
    const int wave = threadIdx.x >> 5;
    const int half = lane >> 4;
    const int ln16 = lane & 15;
    const int m0 = blockIdx.x * 64 + wave * 16;
    const int n0 = blockIdx.y * 64;

    v8f acc[4] = {};
    for (int k = 0; k < KD; k += 32) {
        const _Float16* ap = A + (size_t)(m0 + ln16) * KD + k + half * 8;
        v16h af = cat16(*(const v8h*)ap, *(const v8h*)(ap + 16));
#pragma unroll
        for (int t = 0; t < 4; ++t) {
            const _Float16* bp = Bt + (size_t)(n0 + t * 16 + ln16) * KD + k + half * 8;
            v16h bf = cat16(*(const v8h*)bp, *(const v8h*)(bp + 16));
            acc[t] = wmma_f16(af, bf, acc[t]);
        }
    }
#pragma unroll
    for (int t = 0; t < 4; ++t) {
#pragma unroll
        for (int r = 0; r < 8; ++r) {
            int m  = m0 + half * 8 + r;          // global row (b*SEQ + i)
            int nc = n0 + t * 16 + ln16;         // global col (h*HD + d)
            int b = m >> 11, i = m & (SEQ - 1);
            int h = nc >> 6, d = nc & (HD - 1);
            float v = acc[t][r] * scale;
            size_t addr;
            if (mode == 2) addr = ((size_t)(b * HEADS + h) * HD + d) * SEQ + i;
            else           addr = ((size_t)(b * HEADS + h) * SEQ + i) * HD + d;
            out[addr] = (_Float16)v;
        }
    }
}

// Output projection: out f32 [ROWS][DIM] = ctx[ROWS][INNER] * WoT[DIM][INNER]^T
__global__ __launch_bounds__(128)
void k_outproj(const _Float16* __restrict__ A, const _Float16* __restrict__ Bt,
               float* __restrict__ out) {
    const int lane = threadIdx.x & 31;
    const int wave = threadIdx.x >> 5;
    const int half = lane >> 4;
    const int ln16 = lane & 15;
    const int m0 = blockIdx.x * 64 + wave * 16;
    const int n0 = blockIdx.y * 64;

    v8f acc[4] = {};
    for (int k = 0; k < KD; k += 32) {
        const _Float16* ap = A + (size_t)(m0 + ln16) * KD + k + half * 8;
        v16h af = cat16(*(const v8h*)ap, *(const v8h*)(ap + 16));
#pragma unroll
        for (int t = 0; t < 4; ++t) {
            const _Float16* bp = Bt + (size_t)(n0 + t * 16 + ln16) * KD + k + half * 8;
            v16h bf = cat16(*(const v8h*)bp, *(const v8h*)(bp + 16));
            acc[t] = wmma_f16(af, bf, acc[t]);
        }
    }
#pragma unroll
    for (int t = 0; t < 4; ++t) {
#pragma unroll
        for (int r = 0; r < 8; ++r) {
            int m  = m0 + half * 8 + r;
            int nc = n0 + t * 16 + ln16;
            out[(size_t)m * DIM + nc] = acc[t][r];
        }
    }
}

// ---------------------------------------------------------------------------
// Flash attention. Block = 128 threads (4 waves) = 64 query rows of one (b,h).
// K-tile (32x64) and Vt-tile (64x32) are double-buffered in LDS via async DMA.
// Qh,Kh: [b*h][SEQ][HD] f16 (Q pre-scaled by 1/8) ; Vt: [b*h][HD][SEQ] f16 ;
// ctx: [b][SEQ][INNER] f16
__global__ __launch_bounds__(128)
void k_attn(const _Float16* __restrict__ Qh, const _Float16* __restrict__ Kh,
            const _Float16* __restrict__ Vt, _Float16* __restrict__ ctx) {
    __shared__ __align__(16) _Float16 shK[2][32 * HD];   // [buf][j(32)][d(64)]
    __shared__ __align__(16) _Float16 shV[2][HD * 32];   // [buf][d(64)][j(32)]
    __shared__ __align__(16) _Float16 shP[4][16 * 32];   // per-wave P transpose

    const int tid  = threadIdx.x;
    const int lane = tid & 31;
    const int wave = tid >> 5;
    const int half = lane >> 4;
    const int ln16 = lane & 15;
    const int bh   = blockIdx.x >> 5;                  // / (SEQ/64)
    const int i0   = (blockIdx.x & 31) * 64 + wave * 16;

    const _Float16* Qb = Qh + (size_t)bh * SEQ * HD;
    const _Float16* Kb = Kh + (size_t)bh * SEQ * HD;
    const _Float16* Vb = Vt + (size_t)bh * HD * SEQ;
    _Float16* pl = shP[wave];

    // Q fragments for the two K=32 steps covering d=0..63 (per-wave, kept live)
    v16h aq[2];
#pragma unroll
    for (int s = 0; s < 2; ++s) {
        const _Float16* p = Qb + (size_t)(i0 + ln16) * HD + s * 32 + half * 8;
        aq[s] = cat16(*(const v8h*)p, *(const v8h*)(p + 16));
    }

    // cooperative async stage of one j-block: K 32x64 (256 16B chunks) and
    // V 64x32 (256 16B chunks); 128 threads -> 2+2 instructions each
    auto stage = [&](int buf, int jb) {
#pragma unroll
        for (int c = 0; c < 2; ++c) {
            int kc = tid + c * 128;                    // K chunk 0..255
            int krow = kc >> 3, kcol = (kc & 7) * 8;
            async_copy16(&shK[buf][krow * HD + kcol],
                         Kb + (size_t)(jb + krow) * HD + kcol);
            int vrow = kc >> 2, vcol = (kc & 3) * 8;   // V chunk 0..255
            async_copy16(&shV[buf][vrow * 32 + vcol],
                         Vb + (size_t)vrow * SEQ + jb + vcol);
        }
    };

    stage(0, 0);
    async_wait0();
    __syncthreads();

    v8f O[4] = {};
    float rmax[8], rsum[8];
#pragma unroll
    for (int r = 0; r < 8; ++r) { rmax[r] = -3.0e38f; rsum[r] = 0.0f; }

    for (int jb = 0; jb < SEQ; jb += 32) {
        const int buf = (jb >> 5) & 1;
        if (jb + 32 < SEQ) stage(buf ^ 1, jb + 32);    // prefetch next tile

        // scores for columns jb..jb+31 from the LDS K-tile
        v8f s0 = {}, s1 = {};
#pragma unroll
        for (int s = 0; s < 2; ++s) {
            const _Float16* p0 = &shK[buf][ln16 * HD + s * 32 + half * 8];
            v16h bk0 = cat16(*(const v8h*)p0, *(const v8h*)(p0 + 16));
            s0 = wmma_f16(aq[s], bk0, s0);
            const _Float16* p1 = &shK[buf][(16 + ln16) * HD + s * 32 + half * 8];
            v16h bk1 = cat16(*(const v8h*)p1, *(const v8h*)(p1 + 16));
            s1 = wmma_f16(aq[s], bk1, s1);
        }

        // online softmax: row m = half*8 + r lives in one 16-lane half
#pragma unroll
        for (int r = 0; r < 8; ++r) {
            float v0 = s0[r], v1 = s1[r];
            float mx = fmaxf(v0, v1);
            mx = fmaxf(mx, __shfl_xor(mx, 1, 32));
            mx = fmaxf(mx, __shfl_xor(mx, 2, 32));
            mx = fmaxf(mx, __shfl_xor(mx, 4, 32));
            mx = fmaxf(mx, __shfl_xor(mx, 8, 32));
            float newm  = fmaxf(rmax[r], mx);
            float alpha = __expf(rmax[r] - newm);
            float p0 = __expf(v0 - newm);
            float p1 = __expf(v1 - newm);
            float ps = p0 + p1;
            ps += __shfl_xor(ps, 1, 32);
            ps += __shfl_xor(ps, 2, 32);
            ps += __shfl_xor(ps, 4, 32);
            ps += __shfl_xor(ps, 8, 32);
            rsum[r] = rsum[r] * alpha + ps;
            rmax[r] = newm;
            O[0][r] *= alpha; O[1][r] *= alpha; O[2][r] *= alpha; O[3][r] *= alpha;
            int m = half * 8 + r;                      // C-layout row -> LDS (i,j)
            pl[m * 32 + ln16]      = (_Float16)p0;
            pl[m * 32 + 16 + ln16] = (_Float16)p1;
        }

        // cross-lane RAW through LDS within this wave: wait our DS stores
        asm volatile("s_wait_dscnt 0" ::: "memory");

        // reload P in A-matrix layout (16x32) and do P*V from the LDS V-tile
        const _Float16* pr = pl + ln16 * 32 + half * 8;
        v16h pa = cat16(*(const v8h*)pr, *(const v8h*)(pr + 16));
#pragma unroll
        for (int t = 0; t < 4; ++t) {
            const _Float16* vp = &shV[buf][(t * 16 + ln16) * 32 + half * 8];
            v16h bv = cat16(*(const v8h*)vp, *(const v8h*)(vp + 16));
            O[t] = wmma_f16(pa, bv, O[t]);
        }

        // next tile's DMA must have landed, and everyone must be done reading
        // the current buffers, before the buffers flip
        async_wait0();
        __syncthreads();
    }

    // normalize and write context [b][i][h*64+d]
    const int b = bh >> 4, h = bh & 15;
#pragma unroll
    for (int r = 0; r < 8; ++r) {
        float inv = 1.0f / rsum[r];
        int m = i0 + half * 8 + r;
#pragma unroll
        for (int t = 0; t < 4; ++t) {
            float v = O[t][r] * inv;
            ctx[((size_t)(b * SEQ + m)) * INNER + h * HD + t * 16 + ln16] = (_Float16)v;
        }
    }
}

// ---------------------------------------------------------------------------
extern "C" void kernel_launch(void* const* d_in, const int* in_sizes, int n_in,
                              void* d_out, int out_size, void* d_ws, size_t ws_size,
                              hipStream_t stream) {
    const float* q  = (const float*)d_in[0];
    const float* k  = (const float*)d_in[1];
    const float* v  = (const float*)d_in[2];
    const float* Wq = (const float*)d_in[3];
    const float* Wk = (const float*)d_in[4];
    const float* Wv = (const float*)d_in[5];
    const float* Wo = (const float*)d_in[6];
    float* out = (float*)d_out;

    char* ws = (char*)d_ws;
    size_t off = 0;
    auto alloc = [&](size_t bytes) {
        void* p = ws + off;
        off += (bytes + 255) & ~(size_t)255;
        return p;
    };
    const size_t EL = (size_t)ROWS * DIM;        // 4M elements
    const size_t WE = (size_t)DIM * INNER;       // 1M elements
    _Float16* qf  = (_Float16*)alloc(EL * 2);
    _Float16* kf  = (_Float16*)alloc(EL * 2);
    _Float16* vf  = (_Float16*)alloc(EL * 2);
    _Float16* WqT = (_Float16*)alloc(WE * 2);
    _Float16* WkT = (_Float16*)alloc(WE * 2);
    _Float16* WvT = (_Float16*)alloc(WE * 2);
    _Float16* WoT = (_Float16*)alloc(WE * 2);
    _Float16* Qp  = (_Float16*)alloc(EL * 2);    // [b,h,n,64], pre-scaled
    _Float16* Kp  = (_Float16*)alloc(EL * 2);    // [b,h,n,64]
    _Float16* Vp  = (_Float16*)alloc(EL * 2);    // [b,h,64,n]
    _Float16* ctx = (_Float16*)alloc(EL * 2);    // [b,n,inner]

    const int nEl = (int)EL;
    k_cvt_f16<<<nEl / 256, 256, 0, stream>>>(q, qf, nEl);
    k_cvt_f16<<<nEl / 256, 256, 0, stream>>>(k, kf, nEl);
    k_cvt_f16<<<nEl / 256, 256, 0, stream>>>(v, vf, nEl);

    const int nW = (int)WE;
    k_w_transpose<<<nW / 256, 256, 0, stream>>>(Wq, WqT, DIM, INNER);
    k_w_transpose<<<nW / 256, 256, 0, stream>>>(Wk, WkT, DIM, INNER);
    k_w_transpose<<<nW / 256, 256, 0, stream>>>(Wv, WvT, DIM, INNER);
    k_w_transpose<<<nW / 256, 256, 0, stream>>>(Wo, WoT, INNER, DIM);

    dim3 gp(ROWS / 64, INNER / 64);
    k_proj<<<gp, 128, 0, stream>>>(qf, WqT, Qp, 0.125f, 0);  // * head_dim^-0.5
    k_proj<<<gp, 128, 0, stream>>>(kf, WkT, Kp, 1.0f, 0);
    k_proj<<<gp, 128, 0, stream>>>(vf, WvT, Vp, 1.0f, 2);    // V transposed

    int attn_blocks = BSZ * HEADS * (SEQ / 64);              // 1024 blocks
    k_attn<<<attn_blocks, 128, 0, stream>>>(Qp, Kp, Vp, ctx);

    dim3 go(ROWS / 64, DIM / 64);
    k_outproj<<<go, 128, 0, stream>>>(ctx, WoT, out);
}
